// Transformer_model_76965813944920
// MI455X (gfx1250) — matrix operand order repaired and checked by reading, not compiled, not run on hardware
//
#include <hip/hip_runtime.h>
#include <hip/hip_bf16.h>

// ---------------------------------------------------------------------------
// Graph TransformerConv x2 + classifier for MI455X (gfx1250, wave32, WMMA).
// Memory-bound (~2.3 GB edge gathers/scatters vs ~11 GFLOP dense):
//   - dense linears -> v_wmma_f32_16x16x32_f16, fragments pre-packed in LDS
//     so compute phase uses ds_load_b128 (no scalar u16 gathers)
//   - edge phase    -> warp-per-edge, lane == channel (coalesced 128B rows),
//                      global float atomics for segment max / sum / scatter.
// ---------------------------------------------------------------------------

typedef __attribute__((ext_vector_type(16))) _Float16 v16h;
typedef __attribute__((ext_vector_type(8)))  float    v8f;

#define GN 100000     // nodes  (divisible by 16)
#define GE 1600000    // edges  (divisible by 16)

// ---------------------------------------------------------------------------
// out[M, NCOLS] = A[M, K] @ W[K, NCOLS] + bias.  K in {32,128}, NCOLS in {16,32}.
// Block = 128 threads (4 waves); each wave owns one 16-row tile. M % 16 == 0,
// so the tile-validity guard (row0 < M) is wave-uniform: EXEC stays all-1s
// through the WMMAs and stores need no per-lane predication.
//
// Fragment-major LDS layout (CDNA5 ISA 7.12.2, wave32):
//   A 16x32 f16: lane = g*16 + m (g = lane>>4): a[j] = K=kk+8g+j (j<8),
//                a[8+j] = K=kk+16+8g+j  -> packed as sA[wave][kc][lane][0..15]
//   B 32x16 f16: lane = g*16 + n: b[j] = K=kk+8g+j (j<8), b[8+j] = K=kk+16+8g+j
//                -> packed as sW[kc][nt][lane][0..15]
//   C/D f32:     acc[r] = element (m = 8g + r, col n)
// ---------------------------------------------------------------------------
template <int K, int NCOLS>
__global__ void wmma_gemm_kernel(const float* __restrict__ A,
                                 const float* __restrict__ W,
                                 const float* __restrict__ bias,
                                 float* __restrict__ out, int M)
{
    constexpr int KC = K / 32;       // 32-wide K chunks
    constexpr int NT = NCOLS / 16;   // 16-wide column tiles

    __shared__ alignas(32) _Float16 sA[4][KC][32][16];   // [wave][kc][lane][slot]
    __shared__ alignas(32) _Float16 sW[KC][NT][32][16];  // [kc][nt][lane][slot]

    const int tid  = threadIdx.x;
    const int wave = tid >> 5;
    const int lane = tid & 31;
    const int g    = lane >> 4;
    const int n    = lane & 15;

    // ---- stage W fragments (block-cooperative; W is at most 128x32) ----
    for (int idx = tid; idx < K * NCOLS; idx += 128) {
        const int krow = idx / NCOLS, col = idx - krow * NCOLS;
        const int kc = krow >> 5, kr = krow & 31;
        const int wl = (((kr >> 3) & 1) << 4) | (col & 15);    // lane = g*16 + (col%16)
        sW[kc][col >> 4][wl][(kr & 7) | ((kr >> 4) << 3)] = (_Float16)W[idx];
    }

    const long long row0 = ((long long)blockIdx.x * 4 + wave) * 16;
    const bool active = row0 < (long long)M;             // wave-uniform

    // ---- stage A fragments: float4 coalesced reads, repack to frag order ----
    if (active) {
        for (int idx = lane; idx < 16 * (K / 4); idx += 32) {
            const int r  = idx / (K / 4);
            const int c4 = (idx - r * (K / 4)) * 4;      // c4 % 4 == 0
            const float4 f = *(const float4*)&A[(row0 + r) * K + c4];
            const int kc = c4 >> 5, cc = c4 & 31;
            const int ag   = (cc >> 3) & 1;              // lane group for this run
            const int slot = (cc & 7) | ((cc >> 4) << 3);
            _Float16* p = &sA[wave][kc][(ag << 4) | r][slot];
            p[0] = (_Float16)f.x; p[1] = (_Float16)f.y;
            p[2] = (_Float16)f.z; p[3] = (_Float16)f.w;  // 8B-aligned -> b64 store
        }
    }
    __syncthreads();

    if (!active) return;                                  // whole wave exits

    // ---- compute: vector LDS fragment loads + unrolled WMMAs ----
    v16h afrag[KC];
#pragma unroll
    for (int kc = 0; kc < KC; ++kc)
        afrag[kc] = *(const v16h*)&sA[wave][kc][lane][0]; // 32B -> 2x ds_load_b128

#pragma unroll
    for (int nt = 0; nt < NT; ++nt) {
        v8f acc = {};
#pragma unroll
        for (int kc = 0; kc < KC; ++kc) {
            const v16h b = *(const v16h*)&sW[kc][nt][lane][0];
            acc = __builtin_amdgcn_wmma_f32_16x16x32_f16(
                false, afrag[kc], false, b, (short)0, acc, false, false);
        }
        const float bv = bias[nt * 16 + n];
        float* orow = out + (row0 + 8 * g) * NCOLS + nt * 16 + n;
#pragma unroll
        for (int r = 0; r < 8; ++r)
            orow[(long long)r * NCOLS] = acc[r] + bv;     // no predication needed
    }
}

// --------------------------- edge-phase kernels ----------------------------

__device__ inline void atomicMaxFloat(float* addr, float val) {
    // Works for mixed signs when *addr initialized to -inf.
    if (val >= 0.0f) atomicMax((int*)addr, __float_as_int(val));
    else             atomicMin((unsigned int*)addr, __float_as_uint(val));
}

__global__ void init_ms_kernel(float* __restrict__ m, float* __restrict__ s, int n)
{
    int i = blockIdx.x * blockDim.x + threadIdx.x;
    if (i < n) { m[i] = __int_as_float(0xff800000); s[i] = 0.0f; }
}

// warp-per-edge: lane = channel; alpha = <q[dst], k[src]+e>/sqrt(32); seg-max.
__global__ void edge_alpha_kernel(const float* __restrict__ q,
                                  const float* __restrict__ k,
                                  const float* __restrict__ e,
                                  const long long* __restrict__ src,
                                  const long long* __restrict__ dst,
                                  float* __restrict__ alpha,
                                  float* __restrict__ mmax, int E)
{
    const int lane = threadIdx.x & 31;
    long long eidx = (long long)blockIdx.x * 8 + (threadIdx.x >> 5);
    if (eidx >= (long long)E) return;
    long long sN = src[eidx], dN = dst[eidx];
    float p = q[dN * 32 + lane] * (k[sN * 32 + lane] + e[eidx * 32 + lane]);
#pragma unroll
    for (int off = 16; off > 0; off >>= 1) p += __shfl_xor(p, off, 32);
    if (lane == 0) {
        float a = p * 0.17677669529663687f;  // 1/sqrt(32)
        alpha[eidx] = a;
        atomicMaxFloat(&mmax[dN], a);
    }
}

// thread-per-edge: a = exp(alpha - m[dst]); s[dst] += a.
__global__ void edge_exp_kernel(float* __restrict__ alpha,
                                const float* __restrict__ mmax,
                                float* __restrict__ ssum,
                                const long long* __restrict__ dst, int E)
{
    long long eidx = (long long)blockIdx.x * blockDim.x + threadIdx.x;
    if (eidx < (long long)E) {
        long long dN = dst[eidx];
        float a = expf(alpha[eidx] - mmax[dN]);
        alpha[eidx] = a;
        atomicAdd(&ssum[dN], a);
    }
}

// warp-per-edge: out[dst] += (v[src]+e) * a/(s[dst]+eps).  out pre-init = skip.
__global__ void edge_msg_kernel(const float* __restrict__ v,
                                const float* __restrict__ e,
                                const float* __restrict__ alpha,
                                const float* __restrict__ ssum,
                                const long long* __restrict__ src,
                                const long long* __restrict__ dst,
                                float* __restrict__ out, int E)
{
    const int lane = threadIdx.x & 31;
    long long eidx = (long long)blockIdx.x * 8 + (threadIdx.x >> 5);
    if (eidx >= (long long)E) return;
    long long sN = src[eidx], dN = dst[eidx];
    float w = alpha[eidx] / (ssum[dN] + 1e-16f);
    float msg = (v[sN * 32 + lane] + e[eidx * 32 + lane]) * w;
    atomicAdd(&out[dN * 32 + lane], msg);
}

// row softmax over 16 classes, in place on logits.
__global__ void softmax16_kernel(float* __restrict__ out, int n)
{
    int i = blockIdx.x * blockDim.x + threadIdx.x;
    if (i >= n) return;
    float* row = out + (long long)i * 16;
    float mx = row[0];
#pragma unroll
    for (int c = 1; c < 16; ++c) mx = fmaxf(mx, row[c]);
    float tmp[16], sum = 0.0f;
#pragma unroll
    for (int c = 0; c < 16; ++c) { tmp[c] = expf(row[c] - mx); sum += tmp[c]; }
    float inv = 1.0f / sum;
#pragma unroll
    for (int c = 0; c < 16; ++c) row[c] = tmp[c] * inv;
}

// ------------------------------- host side ---------------------------------

static inline void launch_gemm(const float* A, const float* W, const float* b,
                               float* out, int M, int K, int ncols, hipStream_t st)
{
    const dim3 grd((M + 63) / 64), blk(128);
    if (K == 128 && ncols == 32)
        wmma_gemm_kernel<128, 32><<<grd, blk, 0, st>>>(A, W, b, out, M);
    else if (K == 32 && ncols == 32)
        wmma_gemm_kernel<32, 32><<<grd, blk, 0, st>>>(A, W, b, out, M);
    else
        wmma_gemm_kernel<32, 16><<<grd, blk, 0, st>>>(A, W, b, out, M);
}

// P = {Wq,bq, Wk,bk, Wv,bv, We,be, Wskip,bskip}
static void run_conv(const float* xin, int K, const float* const* P,
                     const long long* src, const long long* dst, const float* ea,
                     float* q, float* kbuf, float* vbuf, float* ebuf,
                     float* alpha, float* mmax, float* ssum,
                     float* hout, hipStream_t st)
{
    launch_gemm(xin, P[0], P[1], q,    GN, K, 32, st);   // q
    launch_gemm(xin, P[2], P[3], kbuf, GN, K, 32, st);   // k
    launch_gemm(xin, P[4], P[5], vbuf, GN, K, 32, st);   // v
    launch_gemm(xin, P[8], P[9], hout, GN, K, 32, st);   // skip -> init output
    launch_gemm(ea,  P[6], P[7], ebuf, GE, 32, 32, st);  // e = edge_attr @ We + be

    init_ms_kernel<<<dim3((GN + 255) / 256), dim3(256), 0, st>>>(mmax, ssum, GN);
    edge_alpha_kernel<<<dim3((GE + 7) / 8), dim3(256), 0, st>>>(
        q, kbuf, ebuf, src, dst, alpha, mmax, GE);
    edge_exp_kernel<<<dim3((GE + 255) / 256), dim3(256), 0, st>>>(
        alpha, mmax, ssum, dst, GE);
    edge_msg_kernel<<<dim3((GE + 7) / 8), dim3(256), 0, st>>>(
        vbuf, ebuf, alpha, ssum, src, dst, hout, GE);
}

extern "C" void kernel_launch(void* const* d_in, const int* in_sizes, int n_in,
                              void* d_out, int out_size, void* d_ws, size_t ws_size,
                              hipStream_t stream)
{
    (void)in_sizes; (void)n_in; (void)out_size; (void)ws_size;

    const float*     x  = (const float*)d_in[0];          // [N,128]
    const long long* ei = (const long long*)d_in[1];      // [2,E] int64
    const float*     ea = (const float*)d_in[2];          // [E,32]
    const long long* src = ei;
    const long long* dst = ei + GE;

    // params in setup_inputs() insertion order:
    // conv1: d_in[3..12], conv2: d_in[13..22], Wl: d_in[23], bl: d_in[24]
    const float* P1[10]; const float* P2[10];
    for (int i = 0; i < 10; ++i) { P1[i] = (const float*)d_in[3 + i];
                                   P2[i] = (const float*)d_in[13 + i]; }
    const float* Wl = (const float*)d_in[23];
    const float* bl = (const float*)d_in[24];

    // workspace carve-out (~277 MB of float scratch)
    float* ws = (float*)d_ws;
    float* q     = ws; ws += (size_t)GN * 32;
    float* kbuf  = ws; ws += (size_t)GN * 32;
    float* vbuf  = ws; ws += (size_t)GN * 32;
    float* hA    = ws; ws += (size_t)GN * 32;
    float* hB    = ws; ws += (size_t)GN * 32;
    float* ebuf  = ws; ws += (size_t)GE * 32;
    float* alpha = ws; ws += (size_t)GE;
    float* mmax  = ws; ws += (size_t)GN;
    float* ssum  = ws; ws += (size_t)GN;

    // layer 1: 128 -> 32
    run_conv(x,  128, P1, src, dst, ea, q, kbuf, vbuf, ebuf, alpha, mmax, ssum, hA, stream);
    // layer 2: 32 -> 32
    run_conv(hA, 32,  P2, src, dst, ea, q, kbuf, vbuf, ebuf, alpha, mmax, ssum, hB, stream);

    // classifier: logits -> d_out, softmax in place
    launch_gemm(hB, Wl, bl, (float*)d_out, GN, 32, 16, stream);
    softmax16_kernel<<<dim3((GN + 255) / 256), dim3(256), 0, stream>>>((float*)d_out, GN);
}
